// EdgeConvSeg_43035572306083
// MI455X (gfx1250) — matrix operand order repaired and checked by reading, not compiled, not run on hardware
//
// EdgeConv (DGCNN) forward for MI455X (gfx1250), wave32 + WMMA.
// All conv1x1/fc layers run through v_wmma_f32_16x16x32_f16 with fused
// bias/BN (y = x*s + t) + ReLU epilogue. Each wave register-blocks a
// (16*MT) x (16*NT) output block: MT*NT WMMAs per K-step share MT A-frags
// and NT B-frags. Output mode is a template param (no epilogue branching).
#include <hip/hip_runtime.h>

#define BDIM 8
#define NPTS 2048
#define KNB  30
#define BN   (BDIM * NPTS)          // 16384 point rows
#define MEDGE (BN * KNB)            // 491520 edge rows

typedef _Float16 h16v __attribute__((ext_vector_type(16)));
typedef _Float16 h8v  __attribute__((ext_vector_type(8)));
typedef float    f8v  __attribute__((ext_vector_type(8)));

// ---------------------------------------------------------------------------
// WMMA GEMM: Out[M x Np] = A[M x Kdim](f16, row-major, lda) *
//                          Bm[Kdim x Np](f16, row-major, ldb)
// Epilogue: y = acc*scale[col] + shift[col]; optional ReLU.
// OM output mode: 0 = f16 row-major; 1 = f16 row-major + f32 row-major;
//                 2 = f32 row-major; 3 = f32 (B,C,N) layout.
// Requires: M%(16*MT)==0, Np%(16*NT)==0, Kdim%32==0, lda%32==0.
// ---------------------------------------------------------------------------
template <int MT, int NT, int OM>
__global__ __launch_bounds__(256) void wmma_gemm_kernel(
    const _Float16* __restrict__ A, int lda,
    const _Float16* __restrict__ Bm, int ldb,
    int M, int Kdim, int Np,
    const float* __restrict__ scale, const float* __restrict__ shift,
    int relu,
    _Float16* __restrict__ out16, int ldo16, int col16,
    float* __restrict__ out32, int ldo32, int bcnN)
{
  const int lane = threadIdx.x & 31;
  const int wave = threadIdx.x >> 5;
  const int mgroups = M / (16 * MT);
  const int ngroups = Np / (16 * NT);
  int tile = blockIdx.x * 8 + wave;
  if (tile >= mgroups * ngroups) return;
  const int ngrp = tile / mgroups;
  const int mgrp = tile - ngrp * mgroups;

  const int g = lane >> 4;                 // K sub-group select
  // A element offsets (keep loads rooted at the global kernel argument so
  // they lower to global_load_b128, not flat_load).
  size_t aoff[MT];
#pragma unroll
  for (int i = 0; i < MT; ++i) {
    int mrow = mgrp * (16 * MT) + i * 16 + (lane & 15);
    aoff[i] = (size_t)mrow * lda + g * 8;
  }
  const _Float16* bp = Bm + (size_t)(g * 8) * ldb + ngrp * (16 * NT) + (lane & 15);
  const size_t bstep = (size_t)32 * ldb;

  f8v acc[MT][NT];
#pragma unroll
  for (int i = 0; i < MT; ++i)
#pragma unroll
    for (int t = 0; t < NT; ++t)
      acc[i][t] = (f8v){0.f,0.f,0.f,0.f,0.f,0.f,0.f,0.f};

  for (int k0 = 0; k0 < Kdim; k0 += 32) {
    // A 16x32 f16 frag: halves 0-7 = K (g?8..15:0..7), halves 8-15 = K+16.
    union { h16v v; h8v h[2]; } ua[MT];
#pragma unroll
    for (int i = 0; i < MT; ++i) {
      ua[i].h[0] = *(const h8v*)(A + aoff[i] + k0);
      ua[i].h[1] = *(const h8v*)(A + aoff[i] + k0 + 16);
    }
    // B 32x16 f16 frags: lane = N column, halves 0-7 = K (g?8..15:0..7), halves 8-15 = K+16. Load all NT
    // fragments first so loads cluster and WMMAs overlap the tail latency.
    h16v bv[NT];
#pragma unroll
    for (int t = 0; t < NT; ++t)
#pragma unroll
      for (int s = 0; s < 16; ++s)
        bv[t][s] = bp[(size_t)(s + (s >> 3) * 8) * ldb + t * 16];
    bp += bstep;
#pragma unroll
    for (int i = 0; i < MT; ++i)
#pragma unroll
      for (int t = 0; t < NT; ++t)
        acc[i][t] = __builtin_amdgcn_wmma_f32_16x16x32_f16(
            false, ua[i].v, false, bv[t], (short)0, acc[i][t], false, false);
  }

#pragma unroll
  for (int i = 0; i < MT; ++i) {
#pragma unroll
    for (int t = 0; t < NT; ++t) {
      const int ncol = ngrp * (16 * NT) + t * 16 + (lane & 15);
      const float s  = scale[ncol];
      const float tt = shift[ncol];
#pragma unroll
      for (int r = 0; r < 8; ++r) {
        float y = acc[i][t][r] * s + tt;
        if (relu) y = y > 0.f ? y : 0.f;
        const int mr = mgrp * (16 * MT) + i * 16 + r + (g << 3);
        if (OM == 0 || OM == 1)
          out16[(size_t)mr * ldo16 + col16 + ncol] = (_Float16)y;
        if (OM == 1 || OM == 2)
          out32[(size_t)mr * ldo32 + ncol] = y;
        if (OM == 3) {
          int b = mr / bcnN, n = mr - b * bcnN;
          out32[((size_t)b * Np + ncol) * bcnN + n] = y;
        }
      }
    }
  }
}

// --------------------------- prep kernels ----------------------------------
// Transpose (Cout x Cin) f32 weight -> (Kp x Np) f16, zero padded.
__global__ void prep_weight_kernel(_Float16* __restrict__ dst,
                                   const float* __restrict__ src,
                                   int Cout, int Cin, int Kp, int Np) {
  int i = blockIdx.x * blockDim.x + threadIdx.x;
  if (i >= Kp * Np) return;
  int r = i / Np, c = i - r * Np;           // r = cin, c = cout
  float v = (r < Cin && c < Cout) ? src[(size_t)c * Cin + r] : 0.f;
  dst[(size_t)r * Np + c] = (_Float16)v;
}

// Per-channel affine: fold conv bias + batchnorm into y = x*s + t.
__global__ void prep_affine_kernel(float* __restrict__ s, float* __restrict__ t,
                                   const float* __restrict__ bias,
                                   const float* __restrict__ g,
                                   const float* __restrict__ bb,
                                   const float* __restrict__ m,
                                   const float* __restrict__ v,
                                   int C, int Np, int hasBN) {
  int c = blockIdx.x * blockDim.x + threadIdx.x;
  if (c >= Np) return;
  if (c >= C) { s[c] = 1.f; t[c] = 0.f; return; }
  if (hasBN) {
    float sc = g[c] * rsqrtf(v[c] + 1e-5f);
    s[c] = sc;
    t[c] = (bias[c] - m[c]) * sc + bb[c];
  } else {
    s[c] = 1.f;
    t[c] = bias[c];
  }
}

// --------------------------- edge features ---------------------------------
// feat: (BN x C) f32 contiguous. ef: (MEDGE x Kp) f16, cols [0,C)=center,
// [C,2C)=nbr-center, rest zero.
__global__ void gather_ef_kernel(_Float16* __restrict__ ef,
                                 const float* __restrict__ feat,
                                 const int* __restrict__ idx,
                                 int C, int Kp) {
  int m = blockIdx.x * blockDim.x + threadIdx.x;
  if (m >= MEDGE) return;
  int bn = m / KNB;
  int b = bn / NPTS;
  int nb = idx[m];
  const float* cen = feat + (size_t)bn * C;
  const float* nbr = feat + ((size_t)b * NPTS + nb) * C;
  _Float16* row = ef + (size_t)m * Kp;
  for (int c = 0; c < C; ++c) {
    float cv = cen[c];
    row[c]     = (_Float16)cv;
    row[C + c] = (_Float16)(nbr[c] - cv);
  }
  for (int c = 2 * C; c < Kp; ++c) row[c] = (_Float16)0.f;
}

// --------------------------- reductions ------------------------------------
// h: (MEDGE x C) f16, edge rows grouped per point. Max (and optional mean)
// over the KNB neighbors -> out[bn*ldo + c] (+ out[bn*ldo + meanofs + c]).
__global__ void reduce_k_kernel(const _Float16* __restrict__ h, int C,
                                _Float16* __restrict__ out, int ldo,
                                int meanofs) {
  int i = blockIdx.x * blockDim.x + threadIdx.x;
  if (i >= BN * C) return;
  int bn = i / C, c = i - bn * C;
  const _Float16* p = h + ((size_t)bn * KNB) * C + c;
  float mx = -3.4e38f, sm = 0.f;
  for (int k = 0; k < KNB; ++k) {
    float v = (float)p[(size_t)k * C];
    mx = v > mx ? v : mx;
    sm += v;
  }
  out[(size_t)bn * ldo + c] = (_Float16)mx;
  if (meanofs >= 0)
    out[(size_t)bn * ldo + meanofs + c] = (_Float16)(sm * (1.f / KNB));
}

// h: (BN x C) f16 -> out: (16 x C) f16, rows b<8 = max over N, rows 8..15 = 0.
__global__ void reduce_n_kernel(const _Float16* __restrict__ h, int C,
                                _Float16* __restrict__ out) {
  int i = blockIdx.x * blockDim.x + threadIdx.x;
  if (i >= 16 * C) return;
  int b = i / C, c = i - b * C;
  if (b >= BDIM) { out[i] = (_Float16)0.f; return; }
  float mx = -3.4e38f;
  const _Float16* p = h + ((size_t)b * NPTS) * C + c;
  for (int n = 0; n < NPTS; ++n) {
    float v = (float)p[(size_t)n * C];
    mx = v > mx ? v : mx;
  }
  out[(size_t)b * C + c] = (_Float16)mx;
}

// --------------------------- transform apply -------------------------------
// tmat: (16 x 16) f32 from fc3 GEMM; row b holds t[c*3+d] in cols 0..8.
__global__ void apply_t_kernel(const float* __restrict__ x,
                               const float* __restrict__ tmat,
                               float* __restrict__ xt) {
  int bn = blockIdx.x * blockDim.x + threadIdx.x;
  if (bn >= BN) return;
  int b = bn / NPTS;
  const float* T = tmat + (size_t)b * 16;
  float p0 = x[(size_t)bn * 3 + 0];
  float p1 = x[(size_t)bn * 3 + 1];
  float p2 = x[(size_t)bn * 3 + 2];
#pragma unroll
  for (int d = 0; d < 3; ++d)
    xt[(size_t)bn * 3 + d] = p0 * T[0 * 3 + d] + p1 * T[1 * 3 + d] + p2 * T[2 * 3 + d];
}

// --------------------------- top-k neighbors -------------------------------
__device__ __forceinline__ void topk_insert(float d, int m, float* dk, int* ik) {
  if (d < dk[KNB - 1]) {
    int j = KNB - 1;
    while (j > 0 && dk[j - 1] > d) { dk[j] = dk[j - 1]; ik[j] = ik[j - 1]; --j; }
    dk[j] = d; ik[j] = m;
  }
}

__global__ void topk3_kernel(const float* __restrict__ feat,
                             int* __restrict__ idx) {
  int bn = blockIdx.x * blockDim.x + threadIdx.x;
  if (bn >= BN) return;
  int b = bn / NPTS;
  const float* fb = feat + (size_t)b * NPTS * 3;
  float p0 = feat[(size_t)bn * 3 + 0];
  float p1 = feat[(size_t)bn * 3 + 1];
  float p2 = feat[(size_t)bn * 3 + 2];
  float dk[KNB]; int ik[KNB];
  for (int j = 0; j < KNB; ++j) { dk[j] = 3.4e38f; ik[j] = 0; }
  for (int m = 0; m < NPTS; ++m) {
    float d0 = p0 - fb[(size_t)m * 3 + 0];
    float d1 = p1 - fb[(size_t)m * 3 + 1];
    float d2 = p2 - fb[(size_t)m * 3 + 2];
    topk_insert(d0 * d0 + d1 * d1 + d2 * d2, m, dk, ik);
  }
  for (int j = 0; j < KNB; ++j) idx[(size_t)bn * KNB + j] = ik[j];
}

__global__ __launch_bounds__(128) void topk64_kernel(
    const float* __restrict__ feat, int* __restrict__ idx) {
  __shared__ float tile[64 * 65];
  const int blocksPerBatch = NPTS / 128;
  int b  = blockIdx.x / blocksPerBatch;
  int n  = (blockIdx.x % blocksPerBatch) * 128 + threadIdx.x;
  int bn = b * NPTS + n;
  const float* fb = feat + (size_t)b * NPTS * 64;
  float my[64];
#pragma unroll
  for (int j = 0; j < 64; ++j) my[j] = fb[(size_t)n * 64 + j];
  float dk[KNB]; int ik[KNB];
  for (int j = 0; j < KNB; ++j) { dk[j] = 3.4e38f; ik[j] = 0; }
  for (int m0 = 0; m0 < NPTS; m0 += 64) {
    __syncthreads();
    if (m0 + 64 < NPTS)
      __builtin_prefetch(fb + (size_t)(m0 + 64) * 64, 0, 0);  // global_prefetch_b8
    for (int t = threadIdx.x; t < 64 * 64; t += 128) {
      int r = t >> 6, c = t & 63;
      tile[r * 65 + c] = fb[(size_t)(m0 + r) * 64 + c];
    }
    __syncthreads();
    for (int r = 0; r < 64; ++r) {
      float d = 0.f;
#pragma unroll
      for (int j = 0; j < 64; ++j) {
        float df = my[j] - tile[r * 65 + j];
        d += df * df;
      }
      topk_insert(d, m0 + r, dk, ik);
    }
  }
  for (int j = 0; j < KNB; ++j) idx[(size_t)bn * KNB + j] = ik[j];
}

// ===========================================================================
extern "C" void kernel_launch(void* const* d_in, const int* in_sizes, int n_in,
                              void* d_out, int out_size, void* d_ws, size_t ws_size,
                              hipStream_t stream) {
  (void)in_sizes; (void)n_in; (void)out_size; (void)ws_size;
  // Input flattening assumption: setup_inputs() dict, depth-first insertion
  // order (bn dicts contribute g,b,m,v; ec's dict.update appends c2 last).
  const float* IN[128];
  for (int i = 0; i < n_in && i < 128; ++i) IN[i] = (const float*)d_in[i];
  const float* x_in = IN[0];

  // ---- workspace bump allocator (256B aligned) ----
  char* ws = (char*)d_ws;
  size_t off = 0;
  auto alloc = [&](size_t bytes) -> void* {
    off = (off + 255) & ~(size_t)255;
    void* p = ws + off;
    off += bytes;
    return p;
  };

  _Float16* bufA    = (_Float16*)alloc((size_t)MEDGE * 128 * 2); // big ping
  _Float16* bufB    = (_Float16*)alloc((size_t)MEDGE * 64 * 2);  // big pong
  _Float16* cat     = (_Float16*)alloc((size_t)BN * 128 * 2);
  _Float16* headcat = (_Float16*)alloc((size_t)BN * 192 * 2);
  _Float16* h3      = (_Float16*)alloc((size_t)BN * 1024 * 2);
  _Float16* gbuf    = (_Float16*)alloc((size_t)16 * 1024 * 2);
  _Float16* fca     = (_Float16*)alloc((size_t)16 * 512 * 2);
  _Float16* fcb     = (_Float16*)alloc((size_t)16 * 256 * 2);
  float*    tmat    = (float*)alloc((size_t)16 * 16 * 4);
  float*    xt      = (float*)alloc((size_t)BN * 3 * 4);
  float*    x1      = (float*)alloc((size_t)BN * 64 * 4);
  float*    x2      = (float*)alloc((size_t)BN * 64 * 4);
  int*      idxb    = (int*)alloc((size_t)BN * KNB * 4);

  // ---- layer table: (Cin, Cout, w idx, bias idx, bn idx or -1) ----
  struct LayerDef { int cin, cout, wi, bi, bni; };
  enum { L_IT1, L_IT2, L_IT3, L_FC1, L_FC2, L_FC3,
         L_E1C1, L_E1C2, L_E1C3, L_E2C1, L_E2C3, L_E3C1, L_E3C3,
         L_HD1, L_HD2, NLAYERS };
  const LayerDef LD[NLAYERS] = {
      {6, 64, 1, 2, 3},      {64, 128, 7, 8, 9},    {128, 1024, 13, 14, 15},
      {1024, 512, 19, 20, 21}, {512, 256, 25, 26, 27}, {256, 9, 31, 32, -1},
      {6, 64, 33, 34, 35},   {64, 64, 45, 46, 47},  {128, 64, 39, 40, 41},
      {128, 64, 51, 52, 53}, {128, 64, 57, 58, 59},
      {128, 64, 63, 64, 65}, {128, 64, 69, 70, 71},
      {192, 640, 75, 76, -1}, {192, 640, 77, 78, -1}};

  _Float16* W[NLAYERS]; float* Sc[NLAYERS]; float* Sh[NLAYERS];
  int Kp[NLAYERS], Np[NLAYERS];
  for (int l = 0; l < NLAYERS; ++l) {
    Kp[l] = (LD[l].cin + 31) & ~31;
    Np[l] = (LD[l].cout + 15) & ~15;
    W[l]  = (_Float16*)alloc((size_t)Kp[l] * Np[l] * 2);
    Sc[l] = (float*)alloc((size_t)Np[l] * 4);
    Sh[l] = (float*)alloc((size_t)Np[l] * 4);
    int nt = Kp[l] * Np[l];
    prep_weight_kernel<<<(nt + 255) / 256, 256, 0, stream>>>(
        W[l], IN[LD[l].wi], LD[l].cout, LD[l].cin, Kp[l], Np[l]);
    const float *g = nullptr, *bb = nullptr, *m = nullptr, *v = nullptr;
    int hasBN = LD[l].bni >= 0;
    if (hasBN) { g = IN[LD[l].bni]; bb = IN[LD[l].bni + 1];
                 m = IN[LD[l].bni + 2]; v = IN[LD[l].bni + 3]; }
    prep_affine_kernel<<<(Np[l] + 255) / 256, 256, 0, stream>>>(
        Sc[l], Sh[l], IN[LD[l].bi], g, bb, m, v, LD[l].cout, Np[l], hasBN);
  }

  // omode: 0 = f16; 1 = f16 + f32 row-major; 2 = f32 row-major; 3 = f32 BCN
  auto gemm = [&](int l, const _Float16* A, int lda, int M, int relu,
                  _Float16* o16, int ldo16, int col16,
                  float* o32, int ldo32, int bcnN, int omode) {
#define GEMM_ARGS A, lda, W[l], Np[l], M, Kp[l], Np[l], Sc[l], Sh[l], relu, \
                  o16, ldo16, col16, o32, ldo32, bcnN
    if (omode == 0) {
      if (M % 32 == 0) {
        int tiles = (M / 32) * (Np[l] / 64);
        wmma_gemm_kernel<2, 4, 0><<<(tiles + 7) / 8, 256, 0, stream>>>(GEMM_ARGS);
      } else {
        int tiles = (M / 16) * (Np[l] / 64);
        wmma_gemm_kernel<1, 4, 0><<<(tiles + 7) / 8, 256, 0, stream>>>(GEMM_ARGS);
      }
    } else if (omode == 1) {
      int tiles = (M / 32) * (Np[l] / 64);
      wmma_gemm_kernel<2, 4, 1><<<(tiles + 7) / 8, 256, 0, stream>>>(GEMM_ARGS);
    } else if (omode == 2) {
      int tiles = (M / 16) * (Np[l] / 16);
      wmma_gemm_kernel<1, 1, 2><<<(tiles + 7) / 8, 256, 0, stream>>>(GEMM_ARGS);
    } else {
      int tiles = (M / 32) * (Np[l] / 64);
      wmma_gemm_kernel<2, 4, 3><<<(tiles + 7) / 8, 256, 0, stream>>>(GEMM_ARGS);
    }
#undef GEMM_ARGS
  };

  const int GE = (MEDGE + 255) / 256;   // per-edge grid
  const int GP = (BN + 255) / 256;      // per-point grid

  // ================= input transform branch =================
  topk3_kernel<<<GP, 256, 0, stream>>>(x_in, idxb);
  gather_ef_kernel<<<GE, 256, 0, stream>>>(bufA, x_in, idxb, 3, 32);
  gemm(L_IT1, bufA, 32, MEDGE, 1, bufB, 64, 0, nullptr, 0, 0, 0);   // 6->64
  gemm(L_IT2, bufB, 64, MEDGE, 1, bufA, 128, 0, nullptr, 0, 0, 0);  // 64->128
  reduce_k_kernel<<<(BN * 128 + 255) / 256, 256, 0, stream>>>(bufA, 128, cat, 128, -1);
  gemm(L_IT3, cat, 128, BN, 1, h3, 1024, 0, nullptr, 0, 0, 0);      // 128->1024
  reduce_n_kernel<<<(16 * 1024 + 255) / 256, 256, 0, stream>>>(h3, 1024, gbuf);
  gemm(L_FC1, gbuf, 1024, 16, 1, fca, 512, 0, nullptr, 0, 0, 0);
  gemm(L_FC2, fca, 512, 16, 1, fcb, 256, 0, nullptr, 0, 0, 0);
  gemm(L_FC3, fcb, 256, 16, 0, nullptr, 0, 0, tmat, 16, 0, 2);      // -> 3x3 (+eye)
  apply_t_kernel<<<GP, 256, 0, stream>>>(x_in, tmat, xt);

  // ================= EdgeConv block 1 (3-dim, two convs) =================
  topk3_kernel<<<GP, 256, 0, stream>>>(xt, idxb);
  gather_ef_kernel<<<GE, 256, 0, stream>>>(bufA, xt, idxb, 3, 32);
  gemm(L_E1C1, bufA, 32, MEDGE, 1, bufB, 64, 0, nullptr, 0, 0, 0);  // 6->64
  gemm(L_E1C2, bufB, 64, MEDGE, 1, bufA, 64, 0, nullptr, 0, 0, 0);  // 64->64
  reduce_k_kernel<<<(BN * 64 + 255) / 256, 256, 0, stream>>>(bufA, 64, cat, 128, 64);
  gemm(L_E1C3, cat, 128, BN, 1, headcat, 192, 0, x1, 64, 0, 1);     // 128->64

  // ================= EdgeConv block 2 (64-dim) =================
  topk64_kernel<<<BDIM * (NPTS / 128), 128, 0, stream>>>(x1, idxb);
  gather_ef_kernel<<<GE, 256, 0, stream>>>(bufA, x1, idxb, 64, 128);
  gemm(L_E2C1, bufA, 128, MEDGE, 1, bufB, 64, 0, nullptr, 0, 0, 0); // 128->64
  reduce_k_kernel<<<(BN * 64 + 255) / 256, 256, 0, stream>>>(bufB, 64, cat, 128, 64);
  gemm(L_E2C3, cat, 128, BN, 1, headcat, 192, 64, x2, 64, 0, 1);

  // ================= EdgeConv block 3 (64-dim) =================
  topk64_kernel<<<BDIM * (NPTS / 128), 128, 0, stream>>>(x2, idxb);
  gather_ef_kernel<<<GE, 256, 0, stream>>>(bufA, x2, idxb, 64, 128);
  gemm(L_E3C1, bufA, 128, MEDGE, 1, bufB, 64, 0, nullptr, 0, 0, 0);
  reduce_k_kernel<<<(BN * 64 + 255) / 256, 256, 0, stream>>>(bufB, 64, cat, 128, 64);
  gemm(L_E3C3, cat, 128, BN, 1, headcat, 192, 128, nullptr, 0, 0, 0);

  // ================= head: two 192->640 convs -> (B,640,N) f32 ===========
  float* out = (float*)d_out;
  gemm(L_HD1, headcat, 192, BN, 0, nullptr, 0, 0, out, 0, NPTS, 3);
  gemm(L_HD2, headcat, 192, BN, 0, nullptr, 0, 0,
       out + (size_t)BDIM * 640 * NPTS, 0, NPTS, 3);
}
